// TrellisLinear_68642167324808
// MI455X (gfx1250) — matrix operand order repaired and checked by reading, not compiled, not run on hardware
//
#include <hip/hip_runtime.h>

#define IN_DIM   4096
#define OUT_DIM  4096
#define NROWS    8192      // B*S = 4*2048
#define NSTEPS   128       // K-tiles in GEMM: 4096/32
#define LSTRIDE  20        // LDS row stride in dwords: 80B = 5*16B, bank-safe

typedef __attribute__((ext_vector_type(16))) __bf16        v16bf;
typedef __attribute__((ext_vector_type(8)))  float         v8f;
typedef __attribute__((ext_vector_type(8)))  unsigned int  v8u;

__device__ __forceinline__ unsigned short f2bf(float f) {
  unsigned u = __float_as_uint(f);
  u += 0x7FFFu + ((u >> 16) & 1u);      // round-to-nearest-even
  return (unsigned short)(u >> 16);
}

// CDNA5 async global->LDS copy (ASYNCcnt); 16B per lane.
__device__ __forceinline__ void async_b128(unsigned lds_off, const void* gaddr) {
  asm volatile("global_load_async_to_lds_b128 %0, %1, off"
               :: "v"(lds_off), "v"(gaddr) : "memory");
}
__device__ __forceinline__ void wait_async0() {
  asm volatile("s_wait_asynccnt 0" ::: "memory");
}

// ---------------------------------------------------------------- stats pass 1
__global__ __launch_bounds__(256) void k_stats_partial(
    const float* __restrict__ x, float* __restrict__ pmin, float* __restrict__ pmax) {
  const int c  = blockIdx.x * 256 + threadIdx.x;   // channel
  const int r0 = blockIdx.y * 128;                 // row chunk
  float mn = 3.402823466e38f, mx = -3.402823466e38f;
  const float* p = x + (size_t)r0 * IN_DIM + c;
  for (int r = 0; r < 128; ++r) {
    float v = p[(size_t)r * IN_DIM];
    mn = fminf(mn, v);
    mx = fmaxf(mx, v);
  }
  pmin[blockIdx.y * IN_DIM + c] = mn;
  pmax[blockIdx.y * IN_DIM + c] = mx;
}

// ---------------------------------------------------------------- stats pass 2
__global__ __launch_bounds__(256) void k_stats_final(
    const float* __restrict__ pmin, const float* __restrict__ pmax,
    float* __restrict__ mn_out, float* __restrict__ sc_out) {
  const int c = blockIdx.x * 256 + threadIdx.x;
  float mn = 3.402823466e38f, mx = -3.402823466e38f;
  for (int b = 0; b < 64; ++b) {
    mn = fminf(mn, pmin[b * IN_DIM + c]);
    mx = fmaxf(mx, pmax[b * IN_DIM + c]);
  }
  float sc = fmaxf((mx - mn) * (1.0f / 255.0f), 1e-8f);
  mn_out[c] = mn;
  sc_out[c] = sc;
}

// ------------------------------------------------------- fake-quant x -> bf16
__global__ __launch_bounds__(256) void k_quant(
    const float* __restrict__ x, const float* __restrict__ mn,
    const float* __restrict__ sc, unsigned short* __restrict__ xq) {
  const size_t e4 = ((size_t)blockIdx.x * 256 + threadIdx.x) * 4;
  const int i0 = (int)(e4 & (IN_DIM - 1));
  float4 v = *(const float4*)(x + e4);
  float m0 = mn[i0], m1 = mn[i0 + 1], m2 = mn[i0 + 2], m3 = mn[i0 + 3];
  float s0 = sc[i0], s1 = sc[i0 + 1], s2 = sc[i0 + 2], s3 = sc[i0 + 3];
  float q0 = fminf(fmaxf(rintf((v.x - m0) / s0), 0.f), 255.f);
  float q1 = fminf(fmaxf(rintf((v.y - m1) / s1), 0.f), 255.f);
  float q2 = fminf(fmaxf(rintf((v.z - m2) / s2), 0.f), 255.f);
  float q3 = fminf(fmaxf(rintf((v.w - m3) / s3), 0.f), 255.f);
  unsigned lo = (unsigned)f2bf(q0 * s0 + m0) | ((unsigned)f2bf(q1 * s1 + m1) << 16);
  unsigned hi = (unsigned)f2bf(q2 * s2 + m2) | ((unsigned)f2bf(q3 * s3 + m3) << 16);
  *(uint2*)(xq + e4) = make_uint2(lo, hi);
}

// --------------------------------- trellis decode + row FWHT (one block/row o)
__global__ __launch_bounds__(256) void k_decode_fwht_row(
    const int* __restrict__ trellis, const float* __restrict__ lut,
    float* __restrict__ s1) {
  __shared__ float row[4096];
  __shared__ int   tw[512];
  const int t = threadIdx.x;
  const int o = blockIdx.x;
  tw[t]       = trellis[(size_t)o * 512 + t];
  tw[t + 256] = trellis[(size_t)o * 512 + 256 + t];
  __syncthreads();
  const int seg   = t >> 4;          // 0..15 trellis sub-row
  const int js    = (t & 15) * 8;    // 8 states per thread
  const int wbase = seg * 32;
  const float2* lut2 = (const float2*)lut;
  #pragma unroll
  for (int jj = 0; jj < 8; ++jj) {
    int j = js + jj;
    int s = 0;
    #pragma unroll
    for (int q = 0; q < 4; ++q) {
      int p   = (j + q) & 127;                              // circular stream
      int nib = (tw[wbase + (p >> 2)] >> (12 - ((p & 3) << 2))) & 15;
      s |= nib << (12 - 4 * q);
    }
    float2 lv = lut2[s];
    row[seg * 256 + 2 * j]     = lv.x;
    row[seg * 256 + 2 * j + 1] = lv.y;
  }
  for (int h = 1; h < 4096; h <<= 1) {
    __syncthreads();
    for (int m = t; m < 2048; m += 256) {
      int i = ((m & ~(h - 1)) << 1) | (m & (h - 1));
      float a = row[i], b = row[i + h];
      row[i] = a + b;
      row[i + h] = a - b;
    }
  }
  __syncthreads();
  for (int i = t; i < 4096; i += 256)
    s1[(size_t)o * IN_DIM + i] = row[i];
}

// ------------- column FWHT + signs + group scales -> bf16 weights [o][i]
__global__ __launch_bounds__(256) void k_fwht_col(
    const float* __restrict__ s1, const float* __restrict__ SU,
    const float* __restrict__ SV, const float* __restrict__ wsc,
    unsigned short* __restrict__ wb) {
  __shared__ float cols[4][4096];   // 64 KB
  const int c0 = blockIdx.x * 4;
  const int t  = threadIdx.x;
  for (int r = t; r < 4096; r += 256) {
    float4 v = *(const float4*)(s1 + (size_t)r * IN_DIM + c0);  // L2-resident
    cols[0][r] = v.x; cols[1][r] = v.y; cols[2][r] = v.z; cols[3][r] = v.w;
  }
  for (int h = 1; h < 4096; h <<= 1) {
    __syncthreads();
    for (int m = t; m < 2048; m += 256) {
      int i = ((m & ~(h - 1)) << 1) | (m & (h - 1));
      #pragma unroll
      for (int c = 0; c < 4; ++c) {
        float a = cols[c][i], b = cols[c][i + h];
        cols[c][i] = a + b;
        cols[c][i + h] = a - b;
      }
    }
  }
  __syncthreads();
  for (int r = t; r < 4096; r += 256) {
    float sv = SV[r] * (1.0f / 4096.0f);   // both orthonormal FWHT norms
    unsigned short h4[4];
    #pragma unroll
    for (int c = 0; c < 4; ++c) {
      int i = c0 + c;
      float v = cols[c][r] * sv * SU[i] * wsc[(r << 4) + (i >> 8)];
      h4[c] = f2bf(v);
    }
    unsigned lo = (unsigned)h4[0] | ((unsigned)h4[1] << 16);
    unsigned hi = (unsigned)h4[2] | ((unsigned)h4[3] << 16);
    *(uint2*)(wb + (size_t)r * IN_DIM + c0) = make_uint2(lo, hi);
  }
}

// -------------------------------------------------- WMMA bf16 GEMM (fp32 acc)
// out[r,o] = sum_i xq[r,i] * w[o,i];  128x128 block tile, BK=32, 8 waves.
// Tiles staged with GLOBAL_LOAD_ASYNC_TO_LDS_B128 (ASYNCcnt), double-buffered.
__global__ __launch_bounds__(256) void k_gemm(
    const unsigned short* __restrict__ xq, const unsigned short* __restrict__ wb,
    float* __restrict__ out) {
  __shared__ unsigned int As[2][128][LSTRIDE];  // bf16 pairs; 80B rows (16B-aligned runs)
  __shared__ unsigned int Bs[2][128][LSTRIDE];

  const int t    = threadIdx.x;
  const int o0   = blockIdx.x * 128;
  const int r0   = blockIdx.y * 128;
  const int row  = t >> 1;        // tile row this thread stages
  const int half = t & 1;         // 16-element half of the 32-wide K slice
  const int lane = t & 31;
  const int wv   = t >> 5;
  const int m_base = (wv & 3) * 32;   // 4 waves down M
  const int n_base = (wv >> 2) * 64;  // 2 waves across N

  // per-thread staging addresses (bytes advance 64 per K-tile)
  const char* gA0 = (const char*)(xq + (size_t)(r0 + row) * IN_DIM + half * 16);
  const char* gB0 = (const char*)(wb + (size_t)(o0 + row) * IN_DIM + half * 16);
  unsigned ldsA[2], ldsB[2];
  ldsA[0] = (unsigned)(size_t)&As[0][row][half * 8];
  ldsA[1] = (unsigned)(size_t)&As[1][row][half * 8];
  ldsB[0] = (unsigned)(size_t)&Bs[0][row][half * 8];
  ldsB[1] = (unsigned)(size_t)&Bs[1][row][half * 8];

  v8f acc[2][4];
  #pragma unroll
  for (int a = 0; a < 2; ++a)
    #pragma unroll
    for (int b = 0; b < 4; ++b)
      #pragma unroll
      for (int d = 0; d < 8; ++d) acc[a][b][d] = 0.0f;

  // prologue: async-stage K-tile 0 into buffer 0
  async_b128(ldsA[0],      gA0);
  async_b128(ldsA[0] + 16, gA0 + 16);
  async_b128(ldsB[0],      gB0);
  async_b128(ldsB[0] + 16, gB0 + 16);
  wait_async0();
  __syncthreads();

  for (int kt = 0; kt < NSTEPS; ++kt) {
    const int cbuf = kt & 1;
    if (kt + 1 < NSTEPS) {
      const int nb = cbuf ^ 1;
      const char* ga = gA0 + (size_t)(kt + 1) * 64;
      const char* gb = gB0 + (size_t)(kt + 1) * 64;
      async_b128(ldsA[nb],      ga);
      async_b128(ldsA[nb] + 16, ga + 16);
      async_b128(ldsB[nb],      gb);
      async_b128(ldsB[nb] + 16, gb + 16);
      if (kt + 4 < NSTEPS) {                      // keep L2 warm ahead
        __builtin_prefetch(gA0 + (size_t)(kt + 4) * 64, 0, 0);
        __builtin_prefetch(gB0 + (size_t)(kt + 4) * 64, 0, 0);
      }
    }

    // Fragments per ISA bf16 layouts; each is two 16B LDS reads.
    v8u aU[2], bU[4];
    {
      const int poffA = (lane >> 4) << 2;   // lanes>=16 hold K+8 (pair +4)
      #pragma unroll
      for (int sm = 0; sm < 2; ++sm) {
        const int m = m_base + sm * 16 + (lane & 15);
        uint4 lo = *(const uint4*)&As[cbuf][m][poffA];      // K pairs v=0..3
        uint4 hi = *(const uint4*)&As[cbuf][m][poffA + 8];  // K pairs v=4..7
        aU[sm][0] = lo.x; aU[sm][1] = lo.y; aU[sm][2] = lo.z; aU[sm][3] = lo.w;
        aU[sm][4] = hi.x; aU[sm][5] = hi.y; aU[sm][6] = hi.z; aU[sm][7] = hi.w;
      }
      const int poffB = (lane >> 4) << 2;   // lanes>=16 hold K+8 (pair +4)
      #pragma unroll
      for (int sn = 0; sn < 4; ++sn) {
        const int n = n_base + sn * 16 + (lane & 15);
        uint4 lo = *(const uint4*)&Bs[cbuf][n][poffB];
        uint4 hi = *(const uint4*)&Bs[cbuf][n][poffB + 8];
        bU[sn][0] = lo.x; bU[sn][1] = lo.y; bU[sn][2] = lo.z; bU[sn][3] = lo.w;
        bU[sn][4] = hi.x; bU[sn][5] = hi.y; bU[sn][6] = hi.z; bU[sn][7] = hi.w;
      }
    }
    #pragma unroll
    for (int sm = 0; sm < 2; ++sm) {
      v16bf av = __builtin_bit_cast(v16bf, aU[sm]);
      #pragma unroll
      for (int sn = 0; sn < 4; ++sn) {
        v16bf bv = __builtin_bit_cast(v16bf, bU[sn]);
        acc[sm][sn] = __builtin_amdgcn_wmma_f32_16x16x32_bf16(
            false, av, false, bv, (short)0, acc[sm][sn], false, false);
      }
    }

    if (kt + 1 < NSTEPS) wait_async0();   // publish next buffer before barrier
    __syncthreads();
  }

  // D layout: VGPR d -> M = d (+8 for lanes 16..31), N = lane&15
  #pragma unroll
  for (int sm = 0; sm < 2; ++sm)
    #pragma unroll
    for (int sn = 0; sn < 4; ++sn) {
      const int n = o0 + n_base + sn * 16 + (lane & 15);
      #pragma unroll
      for (int d = 0; d < 8; ++d) {
        const int m = r0 + m_base + sm * 16 + d + ((lane >> 4) << 3);
        out[(size_t)m * OUT_DIM + n] = acc[sm][sn][d];
      }
    }
}

// ---------------------------------------------------------------------- launch
extern "C" void kernel_launch(void* const* d_in, const int* in_sizes, int n_in,
                              void* d_out, int out_size, void* d_ws, size_t ws_size,
                              hipStream_t stream) {
  const float* x       = (const float*)d_in[0];
  const float* wscales = (const float*)d_in[1];
  const float* SU      = (const float*)d_in[2];
  const float* SV      = (const float*)d_in[3];
  const float* lut     = (const float*)d_in[4];
  const int*   trellis = (const int*)d_in[5];
  float*       out     = (float*)d_out;

  char* ws = (char*)d_ws;
  float* pmin          = (float*)(ws);                                // 1 MB
  float* pmax          = (float*)(ws + ((size_t)1 << 20));            // 1 MB
  float* mn            = (float*)(ws + ((size_t)2 << 20));            // 16 KB
  float* sc            = (float*)(ws + ((size_t)2 << 20) + 16384);    // 16 KB
  unsigned short* xq   = (unsigned short*)(ws + ((size_t)4 << 20));   // 64 MB
  float* s1            = (float*)(ws + ((size_t)68 << 20));           // 64 MB
  unsigned short* wb   = (unsigned short*)(ws + ((size_t)132 << 20)); // 32 MB

  k_stats_partial<<<dim3(16, 64), 256, 0, stream>>>(x, pmin, pmax);
  k_stats_final<<<16, 256, 0, stream>>>(pmin, pmax, mn, sc);
  k_quant<<<(NROWS * IN_DIM) / (4 * 256), 256, 0, stream>>>(x, mn, sc, xq);
  k_decode_fwht_row<<<OUT_DIM, 256, 0, stream>>>(trellis, lut, s1);
  k_fwht_col<<<IN_DIM / 4, 256, 0, stream>>>(s1, SU, SV, wscales, wb);
  k_gemm<<<dim3(OUT_DIM / 128, NROWS / 128), 256, 0, stream>>>(xq, wb, out);
}
